// EIGNN_21019569947065
// MI455X (gfx1250) — matrix operand order repaired and checked by reading, not compiled, not run on hardware
//
#include <hip/hip_runtime.h>

typedef __attribute__((ext_vector_type(16))) _Float16 v16h;
typedef __attribute__((ext_vector_type(8)))  _Float16 v8h;
typedef __attribute__((ext_vector_type(8)))  float    v8f;

#define NN   8192
#define DIN  64
#define DOUT 32

// ---------------------------------------------------------------------------
// Kernel 1: gF = F^T F / (||F^T F||_F + 1e-6)   (one workgroup, 64x64)
// ---------------------------------------------------------------------------
__global__ __launch_bounds__(256) void k_gf(const float* __restrict__ F,
                                            float* __restrict__ gF) {
  __shared__ float sF[DIN * DIN];
  __shared__ float part[256];
  int t = threadIdx.x;
  for (int i = 0; i < 16; ++i) sF[t + 256 * i] = F[t + 256 * i];
  __syncthreads();
  float v[16];
  float ssq = 0.f;
  for (int i = 0; i < 16; ++i) {
    int idx = t + 256 * i;
    int r = idx >> 6, c = idx & 63;
    float acc = 0.f;
    for (int k = 0; k < 64; ++k) acc += sF[k * 64 + r] * sF[k * 64 + c];
    v[i] = acc;
    ssq += acc * acc;
  }
  part[t] = ssq;
  __syncthreads();
  for (int s = 128; s > 0; s >>= 1) {
    if (t < s) part[t] += part[t + s];
    __syncthreads();
  }
  float inv = 1.f / (sqrtf(part[0]) + 1e-6f);
  for (int i = 0; i < 16; ++i) gF[t + 256 * i] = v[i] * inv;
}

// ---------------------------------------------------------------------------
// Kernel 2: parallel cyclic Jacobi eigensolver for the 64x64 symmetric gF.
// Tournament schedule: 63 rounds x 32 disjoint pairs; rotations on disjoint
// pairs commute, so every round is exact. 15 sweeps (fixed -> deterministic).
// Q columns = eigenvectors, lam[i] = A[i][i] (matches eigh up to ordering,
// which cancels since Q_F appears on both sides with matching lam order).
// ---------------------------------------------------------------------------
__global__ __launch_bounds__(256) void k_jacobi(const float* __restrict__ gF,
                                                float* __restrict__ QF,
                                                float* __restrict__ lamF) {
  __shared__ float A[64 * 64];
  __shared__ float Q[64 * 64];
  __shared__ float rc[32], rs[32];
  __shared__ int   rp[32], rq[32];
  __shared__ int   perm[64];
  int t = threadIdx.x;
  for (int i = 0; i < 16; ++i) {
    int idx = t + 256 * i;
    A[idx] = gF[idx];
    Q[idx] = ((idx >> 6) == (idx & 63)) ? 1.f : 0.f;
  }
  if (t < 64) perm[t] = t;
  __syncthreads();
  for (int sweep = 0; sweep < 15; ++sweep) {
    for (int round = 0; round < 63; ++round) {
      if (t < 32) {
        int a = perm[t], b = perm[63 - t];
        int p = a < b ? a : b;
        int q = a < b ? b : a;
        rp[t] = p; rq[t] = q;
        float app = A[p * 64 + p], aqq = A[q * 64 + q], apq = A[p * 64 + q];
        float c = 1.f, s = 0.f;
        if (fabsf(apq) > 1e-12f) {
          float tau = (aqq - app) / (2.f * apq);
          float tt = (tau >= 0.f ? 1.f : -1.f) / (fabsf(tau) + sqrtf(1.f + tau * tau));
          c = 1.f / sqrtf(1.f + tt * tt);
          s = tt * c;
        }
        rc[t] = c; rs[t] = s;
      }
      __syncthreads();
      // Row rotations: A <- J^T A   (rows of distinct pairs are disjoint)
      for (int it = 0; it < 8; ++it) {
        int idx = t + 256 * it;           // 32 pairs * 64 cols
        int j = idx >> 6, k = idx & 63;
        int p = rp[j], q = rq[j];
        float c = rc[j], s = rs[j];
        float apk = A[p * 64 + k], aqk = A[q * 64 + k];
        A[p * 64 + k] = c * apk - s * aqk;
        A[q * 64 + k] = s * apk + c * aqk;
      }
      __syncthreads();
      // Column rotations: A <- A J ; accumulate Q <- Q J
      for (int it = 0; it < 8; ++it) {
        int idx = t + 256 * it;
        int j = idx >> 6, k = idx & 63;
        int p = rp[j], q = rq[j];
        float c = rc[j], s = rs[j];
        float akp = A[k * 64 + p], akq = A[k * 64 + q];
        A[k * 64 + p] = c * akp - s * akq;
        A[k * 64 + q] = s * akp + c * akq;
        float qkp = Q[k * 64 + p], qkq = Q[k * 64 + q];
        Q[k * 64 + p] = c * qkp - s * qkq;
        Q[k * 64 + q] = s * qkp + c * qkq;
      }
      __syncthreads();
      if (t == 0) {                       // rotate tournament schedule
        int last = perm[63];
        for (int i = 63; i > 1; --i) perm[i] = perm[i - 1];
        perm[1] = last;
      }
      __syncthreads();
    }
  }
  for (int i = 0; i < 16; ++i) { int idx = t + 256 * i; QF[idx] = Q[idx]; }
  if (t < 64) lamF[t] = A[t * 64 + t];
}

// ---------------------------------------------------------------------------
// Kernel 3: W1[i][n] = sum_d QF[d][i] * x[n][d]   -> f16 [64][8192]
// ---------------------------------------------------------------------------
__global__ __launch_bounds__(256) void k_w1(const float* __restrict__ x,
                                            const float* __restrict__ QF,
                                            _Float16* __restrict__ W1h) {
  __shared__ float sQ[64 * 64];
  __shared__ float sx[4][64];
  int t = threadIdx.x, tx = t & 63, ty = t >> 6;
  for (int i = 0; i < 16; ++i) sQ[t + 256 * i] = QF[t + 256 * i];
  int n0 = blockIdx.x * 4;
  sx[ty][tx] = x[(size_t)(n0 + ty) * 64 + tx];
  __syncthreads();
  float acc = 0.f;
  for (int d = 0; d < 64; ++d) acc += sQ[d * 64 + tx] * sx[ty][d];
  W1h[(size_t)tx * NN + n0 + ty] = (_Float16)acc;
}

// ---------------------------------------------------------------------------
// Kernel 4: T1g = G .* (W1 @ Q_S)   (the first big GEMM, fused spectral filter)
// 256 WGs x 8 waves; WG owns 32 cols of Q_S; wave = (col-group g, K-slice s).
// Per 32-K chunk each wave issues 4x v_wmma_f32_16x16x32_f16 (M=64 tile).
// K-slice partials reduced in LDS in a fixed order (deterministic).
// ---------------------------------------------------------------------------
__global__ __launch_bounds__(256) void k_gemm1(
    const _Float16* __restrict__ W1h, const float* __restrict__ QS,
    const float* __restrict__ lamF, const float* __restrict__ lamS,
    const float* __restrict__ gptr, float* __restrict__ T1g) {
  __shared__ float red[8][64 * 16];
  int t = threadIdx.x;
  int lane = t & 31, w = t >> 5;
  int g = w & 1, sl = w >> 1;
  int n0 = blockIdx.x * 32;
  int col = n0 + g * 16 + (lane & 15);
  int hilo = lane >> 4;
  v8f acc[4] = {};
  int kbeg = sl * 2048, kend = kbeg + 2048;
  for (int kb = kbeg; kb < kend; kb += 32) {
    // B fragment (32x16 f16): lane holds N=lane%16, K = (lane<16?0:16)+h
    v16h b;
    {
      int k0 = kb + (hilo ? 16 : 0);
      const float* bp = QS + (size_t)k0 * NN + col;
      __builtin_prefetch(bp + (size_t)32 * NN, 0, 1);  // next chunk, same col
#pragma unroll
      for (int h = 0; h < 16; ++h) b[h] = (_Float16)bp[(size_t)h * NN];
    }
    int abase = kb + hilo * 16;
#pragma unroll
    for (int mg = 0; mg < 4; ++mg) {
      // A fragment (16x32 f16): lane holds M=lane%16; halves 0..7 -> K=base..,
      // halves 8..15 -> K=base+8..  (base = lane<16 ? 0 : 16)
      const _Float16* ap = W1h + (size_t)(mg * 16 + (lane & 15)) * NN + abase;
      v8h c0 = *(const v8h*)(ap);
      v8h c1 = *(const v8h*)(ap + 8);
      v16h a;
#pragma unroll
      for (int i = 0; i < 8; ++i) { a[i] = c0[i]; a[8 + i] = c1[i]; }
      acc[mg] = __builtin_amdgcn_wmma_f32_16x16x32_f16(
          false, a, false, b, (short)0, acc[mg], false, false);
    }
  }
  // C layout: lane L, reg r -> M = r + (L<16?0:8), N = L%16
#pragma unroll
  for (int mg = 0; mg < 4; ++mg)
#pragma unroll
    for (int r = 0; r < 8; ++r)
      red[w][(mg * 16 + r + hilo * 8) * 16 + (lane & 15)] = acc[mg][r];
  __syncthreads();
  float gamma = gptr[0];
  for (int it = 0; it < 8; ++it) {
    int idx = t + 256 * it;                 // 2 groups * 64 m * 16 n
    int gg = idx >> 10, rem = idx & 1023, m = rem >> 4, nc = rem & 15;
    float sum = red[0 + gg][m * 16 + nc] + red[2 + gg][m * 16 + nc] +
                red[4 + gg][m * 16 + nc] + red[6 + gg][m * 16 + nc];
    int j = n0 + gg * 16 + nc;
    float scale = 1.f / (1.f - gamma * lamF[m] * lamS[j]);
    T1g[(size_t)m * NN + j] = sum * scale;
  }
}

// ---------------------------------------------------------------------------
// Kernel 5: W2[i][n] = sum_d QF[i][d] * T1g[d][n]  -> f16 [64][8192]
// ---------------------------------------------------------------------------
__global__ __launch_bounds__(256) void k_w2(const float* __restrict__ T1g,
                                            const float* __restrict__ QF,
                                            _Float16* __restrict__ W2h) {
  __shared__ float sQt[64 * 64];   // transposed: sQt[d*64+i] = QF[i][d]
  __shared__ float sT[4][64];
  int t = threadIdx.x, tx = t & 63, ty = t >> 6;
  for (int i = 0; i < 16; ++i) {
    int idx = t + 256 * i;
    sQt[(idx & 63) * 64 + (idx >> 6)] = QF[idx];
  }
  int n0 = blockIdx.x * 4;
  sT[ty][tx] = T1g[(size_t)tx * NN + n0 + ty];
  __syncthreads();
  float acc = 0.f;
  for (int d = 0; d < 64; ++d) acc += sQt[d * 64 + tx] * sT[ty][d];
  W2h[(size_t)tx * NN + n0 + ty] = (_Float16)acc;
}

// ---------------------------------------------------------------------------
// Kernel 6: Z = W2 @ Q_S^T, fused with out[n][o] = sum_i Z[i][n]*B_w[o][i].
// B[k][n] = Q_S[n][k]: each lane streams 64B contiguous along a Q_S row.
// ---------------------------------------------------------------------------
__global__ __launch_bounds__(256) void k_gemm2(
    const _Float16* __restrict__ W2h, const float* __restrict__ QS,
    const float* __restrict__ Bw, float* __restrict__ out) {
  __shared__ float red[8][64 * 16];
  __shared__ float zt[64][32];
  __shared__ float sBw[DOUT * DIN];
  int t = threadIdx.x;
  for (int i = 0; i < 8; ++i) sBw[t + 256 * i] = Bw[t + 256 * i];
  int lane = t & 31, w = t >> 5;
  int g = w & 1, sl = w >> 1;
  int n0 = blockIdx.x * 32;
  int row = n0 + g * 16 + (lane & 15);      // output n index == Q_S row
  int hilo = lane >> 4;
  v8f acc[4] = {};
  int kbeg = sl * 2048, kend = kbeg + 2048;
  for (int kb = kbeg; kb < kend; kb += 32) {
    v16h b;
    {
      int k0 = kb + (hilo ? 16 : 0);
      const float* bp = QS + (size_t)row * NN + k0;
      __builtin_prefetch(bp + 32, 0, 1);
      float4 f0 = *(const float4*)(bp);
      float4 f1 = *(const float4*)(bp + 4);
      float4 f2 = *(const float4*)(bp + 8);
      float4 f3 = *(const float4*)(bp + 12);
      b[0]  = (_Float16)f0.x; b[1]  = (_Float16)f0.y;
      b[2]  = (_Float16)f0.z; b[3]  = (_Float16)f0.w;
      b[4]  = (_Float16)f1.x; b[5]  = (_Float16)f1.y;
      b[6]  = (_Float16)f1.z; b[7]  = (_Float16)f1.w;
      b[8]  = (_Float16)f2.x; b[9]  = (_Float16)f2.y;
      b[10] = (_Float16)f2.z; b[11] = (_Float16)f2.w;
      b[12] = (_Float16)f3.x; b[13] = (_Float16)f3.y;
      b[14] = (_Float16)f3.z; b[15] = (_Float16)f3.w;
    }
    int abase = kb + hilo * 16;
#pragma unroll
    for (int mg = 0; mg < 4; ++mg) {
      const _Float16* ap = W2h + (size_t)(mg * 16 + (lane & 15)) * NN + abase;
      v8h c0 = *(const v8h*)(ap);
      v8h c1 = *(const v8h*)(ap + 8);
      v16h a;
#pragma unroll
      for (int i = 0; i < 8; ++i) { a[i] = c0[i]; a[8 + i] = c1[i]; }
      acc[mg] = __builtin_amdgcn_wmma_f32_16x16x32_f16(
          false, a, false, b, (short)0, acc[mg], false, false);
    }
  }
#pragma unroll
  for (int mg = 0; mg < 4; ++mg)
#pragma unroll
    for (int r = 0; r < 8; ++r)
      red[w][(mg * 16 + r + hilo * 8) * 16 + (lane & 15)] = acc[mg][r];
  __syncthreads();
  for (int it = 0; it < 8; ++it) {
    int idx = t + 256 * it;
    int gg = idx >> 10, rem = idx & 1023, m = rem >> 4, nc = rem & 15;
    float sum = red[0 + gg][m * 16 + nc] + red[2 + gg][m * 16 + nc] +
                red[4 + gg][m * 16 + nc] + red[6 + gg][m * 16 + nc];
    zt[m][gg * 16 + nc] = sum;
  }
  __syncthreads();
  for (int it = 0; it < 4; ++it) {
    int idx = t + 256 * it;                 // 32 n * 32 o
    int nl = idx >> 5, o = idx & 31;
    float a2 = 0.f;
    for (int i = 0; i < 64; ++i) a2 += zt[i][nl] * sBw[o * 64 + i];
    out[(size_t)(n0 + nl) * DOUT + o] = a2;
  }
}

// ---------------------------------------------------------------------------
extern "C" void kernel_launch(void* const* d_in, const int* in_sizes, int n_in,
                              void* d_out, int out_size, void* d_ws, size_t ws_size,
                              hipStream_t stream) {
  (void)in_sizes; (void)n_in; (void)out_size; (void)ws_size;
  const float* x    = (const float*)d_in[0];   // [8192,64]
  const float* QS   = (const float*)d_in[1];   // [8192,8192]
  const float* lamS = (const float*)d_in[2];   // [8192]
  const float* F    = (const float*)d_in[3];   // [64,64]
  const float* Bw   = (const float*)d_in[4];   // [32,64]
  const float* gmm  = (const float*)d_in[5];   // scalar
  float* out = (float*)d_out;                  // [8192,32]

  // Workspace layout (floats): total 1,056,896 floats ~= 4.03 MiB
  float* wsf  = (float*)d_ws;
  float* gF   = wsf;                                   //  4096
  float* QF   = wsf + 4096;                            //  4096
  float* lamF = wsf + 8192;                            //    64 (+pad to 8320)
  _Float16* W1h = (_Float16*)(wsf + 8320);             // 64*8192 halves
  float*    T1g = wsf + 8320 + 262144;                 // 64*8192 floats
  _Float16* W2h = (_Float16*)(wsf + 8320 + 262144 + 524288);

  k_gf    <<<1,    256, 0, stream>>>(F, gF);
  k_jacobi<<<1,    256, 0, stream>>>(gF, QF, lamF);
  k_w1    <<<2048, 256, 0, stream>>>(x, QF, W1h);
  k_gemm1 <<<256,  256, 0, stream>>>(W1h, QS, lamF, lamS, gmm, T1g);
  k_w2    <<<2048, 256, 0, stream>>>(T1g, QF, W2h);
  k_gemm2 <<<256,  256, 0, stream>>>(W2h, QS, Bw, out);
}
